// ObjectNeck_38001870635911
// MI455X (gfx1250) — compile-verified, matrix operand order checked by reading, not run on hardware
//
#include <hip/hip_runtime.h>
#include <hip/hip_bf16.h>
#include <cstdint>

// ---------------- problem constants ----------------
#define BSZ   64
#define CCH   2048
#define LL    196          // H*W
#define LP    224          // padded L (multiple of 32 so it can be a WMMA K dim)
#define HID   2048
#define OUTC  256
#define NH    8
#define KD    32           // OUT / NH  (attention key count)
#define BN_EPS 1e-5f
#define NORM_EPS 1e-12f

typedef __bf16 bhalf;
typedef __attribute__((ext_vector_type(16))) __bf16 v16bf;
typedef __attribute__((ext_vector_type(8)))  __bf16 v8bf;
typedef __attribute__((ext_vector_type(8)))  float  v8f;

union FragU { v16bf v; v8bf h[2]; };

// Load a 16x32 bf16 WMMA A/B fragment from a K-major matrix (row-major [rows][ld]).
// Per ISA layout: lane&15 selects the row, lane>>4 selects which 8-wide K half,
// giving two contiguous 16B chunks per lane -> two global_load_b128.
__device__ __forceinline__ v16bf load_frag(const bhalf* __restrict__ p, int ld,
                                           int row0, int k0, int lane) {
    int r    = row0 + (lane & 15);
    int half = (lane >> 4) & 1;
    const bhalf* base = p + (size_t)r * ld + k0 + half * 8;
    FragU f;
    f.h[0] = *(const v8bf*)(base);        // K = k0 + half*8 + [0..7]
    f.h[1] = *(const v8bf*)(base + 16);   // K = k0 + 16 + half*8 + [0..7]
    return f.v;
}

// MFx NF register-tiled bf16 WMMA mainloop, one wave per tile (tile = 16*MF x 16*NF).
template<int MF, int NF>
__device__ __forceinline__ void wmma_mainloop(const bhalf* __restrict__ A, int lda, int m0,
                                              const bhalf* __restrict__ B, int ldb, int n0,
                                              int K, v8f acc[MF][NF]) {
    const int lane = threadIdx.x & 31;
#pragma unroll
    for (int i = 0; i < MF; ++i)
#pragma unroll
        for (int j = 0; j < NF; ++j)
#pragma unroll
            for (int r = 0; r < 8; ++r) acc[i][j][r] = 0.0f;

    for (int k0 = 0; k0 < K; k0 += 32) {
        v16bf a[MF], b[NF];
#pragma unroll
        for (int i = 0; i < MF; ++i) a[i] = load_frag(A, lda, m0 + 16 * i, k0, lane);
#pragma unroll
        for (int j = 0; j < NF; ++j) b[j] = load_frag(B, ldb, n0 + 16 * j, k0, lane);
        if (k0 + 32 < K) {  // global_prefetch_b8 of next A K-slice
            __builtin_prefetch(A + (size_t)(m0 + (lane & 15)) * lda + k0 + 32, 0, 1);
        }
#pragma unroll
        for (int i = 0; i < MF; ++i)
#pragma unroll
            for (int j = 0; j < NF; ++j)
                acc[i][j] = __builtin_amdgcn_wmma_f32_16x16x32_bf16(
                    false, a[i], false, b[j], (short)0, acc[i][j], false, false);
    }
}

// D layout: lane&15 = N column, VGPR r = M = 16*i + 8*(lane>>4) + r  -> 8 consecutive M.
template<int MF, int NF>
__device__ __forceinline__ void store_t_f32(v8f acc[MF][NF], float* __restrict__ D,
                                            int ldd, int m0, int n0) {
    const int lane = threadIdx.x & 31, half = lane >> 4;
#pragma unroll
    for (int j = 0; j < NF; ++j) {
        int n = n0 + 16 * j + (lane & 15);
#pragma unroll
        for (int i = 0; i < MF; ++i) {
            float* p = D + (size_t)n * ldd + (m0 + 16 * i + 8 * half);
#pragma unroll
            for (int r = 0; r < 8; ++r) p[r] = acc[i][j][r];
        }
    }
}

template<int MF, int NF>
__device__ __forceinline__ void store_t_bf16(v8f acc[MF][NF], bhalf* __restrict__ D,
                                             int ldd, int m0, int n0) {
    const int lane = threadIdx.x & 31, half = lane >> 4;
#pragma unroll
    for (int j = 0; j < NF; ++j) {
        int n = n0 + 16 * j + (lane & 15);
#pragma unroll
        for (int i = 0; i < MF; ++i) {
            bhalf* p = D + (size_t)n * ldd + (m0 + 16 * i + 8 * half);
#pragma unroll
            for (int r = 0; r < 8; ++r) p[r] = (bhalf)acc[i][j][r];
        }
    }
}

template<int MF, int NF>
__device__ __forceinline__ void store_n_f32(v8f acc[MF][NF], float* __restrict__ D,
                                            int ldd, int m0, int n0) {
    const int lane = threadIdx.x & 31, half = lane >> 4;
#pragma unroll
    for (int j = 0; j < NF; ++j) {
        int n = n0 + 16 * j + (lane & 15);
#pragma unroll
        for (int i = 0; i < MF; ++i)
#pragma unroll
            for (int r = 0; r < 8; ++r)
                D[(size_t)(m0 + 16 * i + 8 * half + r) * ldd + n] = acc[i][j][r];
    }
}

// ---------------- elementwise / prep kernels ----------------

__global__ void k_f32_to_bf16(const float* __restrict__ in, bhalf* __restrict__ out, int n) {
    int i = blockIdx.x * blockDim.x + threadIdx.x;
    if (i < n) out[i] = (bhalf)in[i];
}

__global__ void k_zero(float* __restrict__ p, int n) {
    int i = blockIdx.x * blockDim.x + threadIdx.x;
    if (i < n) p[i] = 0.0f;
}

// One wave per (b,c) row: mean-pool over L, write bf16 row of xh_b (padded to LP).
__global__ void k_pool_xhb(const float* __restrict__ x, float* __restrict__ pool,
                           bhalf* __restrict__ xh_b) {
    int wid  = blockIdx.x * (blockDim.x / 32) + (threadIdx.x / 32);
    int lane = threadIdx.x & 31;
    if (wid >= BSZ * CCH) return;
    const float* row = x + (size_t)wid * LL;
    bhalf* orow = xh_b + (size_t)wid * LP;
    float s = 0.0f;
    for (int l = lane; l < LP; l += 32) {
        float v = (l < LL) ? row[l] : 0.0f;
        s += v;
        orow[l] = (bhalf)v;
    }
    for (int off = 16; off > 0; off >>= 1) s += __shfl_down(s, off, 32);
    if (lane == 0) pool[wid] = s * (1.0f / LL);
}

// Tiled transpose: xcat_t[b][lt][c] (bf16), lt=0 -> pool, lt=1..196 -> x, rest zero.
__global__ void k_xcat_t(const float* __restrict__ x, const float* __restrict__ pool,
                         bhalf* __restrict__ xcat_t) {
    __shared__ float tile[32][33];
    int b = blockIdx.z, lt0 = blockIdx.x * 32, c0 = blockIdx.y * 32;
    for (int yy = threadIdx.y; yy < 32; yy += 8) {
        int c = c0 + yy, lt = lt0 + threadIdx.x;
        float v;
        if (lt == 0)        v = pool[b * CCH + c];
        else if (lt <= LL)  v = x[((size_t)b * CCH + c) * LL + (lt - 1)];
        else                v = 0.0f;
        tile[yy][threadIdx.x] = v;
    }
    __syncthreads();
    for (int yy = threadIdx.y; yy < 32; yy += 8) {
        int lt = lt0 + yy, c = c0 + threadIdx.x;
        xcat_t[((size_t)b * LP + lt) * CCH + c] = (bhalf)tile[threadIdx.x][yy];
    }
}

// Per-channel sum / sumsq over (outer, l<lValid) of a [outer][lPitch][2048] f32 tensor.
__global__ void k_bnreduce(const float* __restrict__ Hm, int lValid, int lPitch,
                           float* __restrict__ sum, float* __restrict__ sq) {
    int c = blockIdx.x * blockDim.x + threadIdx.x;   // grid.x*256 == 2048
    int o = blockIdx.y;
    float s = 0.0f, q = 0.0f;
    for (int l = 0; l < lValid; ++l) {
        float v = Hm[((size_t)o * lPitch + l) * 2048 + c];
        s += v; q += v * v;
    }
    atomicAdd(&sum[c], s);
    atomicAdd(&sq[c], q);
}

// BN (affine) + ReLU + bf16 convert; innermost dim == channel (2048).
__global__ void k_bnrelu(const float* __restrict__ Hm, const float* __restrict__ sum,
                         const float* __restrict__ sq, const float* __restrict__ gamma,
                         const float* __restrict__ beta, float invCnt, int total,
                         bhalf* __restrict__ out) {
    int i = blockIdx.x * blockDim.x + threadIdx.x;
    if (i >= total) return;
    int c = i & 2047;
    float mu  = sum[c] * invCnt;
    float var = sq[c] * invCnt - mu * mu;
    float sc  = gamma[c] * rsqrtf(var + BN_EPS);
    float sh  = beta[c] - mu * sc;
    float v   = Hm[i] * sc + sh;
    out[i] = (bhalf)fmaxf(v, 0.0f);
}

// z_g = z[:, :, 0]
__global__ void k_zg(const float* __restrict__ z, float* __restrict__ out) {
    int i = blockIdx.x * blockDim.x + threadIdx.x;   // b*256 + o
    if (i < BSZ * OUTC) out[i] = z[(size_t)i * LP];
}

// inv_nrm[b][l] = 1 / max(sqrt(sum_c z^2), NORM_EPS); z layout [b][256][LP]
__global__ void k_norm(const float* __restrict__ z, float* __restrict__ inv_nrm) {
    __shared__ float red[8][33];
    int b = blockIdx.y, l = blockIdx.x * 32 + threadIdx.x;
    float s = 0.0f;
    for (int c = threadIdx.y; c < OUTC; c += 8) {
        float v = z[((size_t)b * OUTC + c) * LP + l];
        s += v * v;
    }
    red[threadIdx.y][threadIdx.x] = s;
    __syncthreads();
    if (threadIdx.y == 0) {
        float t = 0.0f;
        for (int k = 0; k < 8; ++k) t += red[k][threadIdx.x];
        inv_nrm[b * LP + l] = 1.0f / fmaxf(sqrtf(t), NORM_EPS);
    }
}

// Softmax over L per (b,o2) row; writes fp32 attn twice to d_out and bf16 padded copy.
__global__ void k_softmax(const float* __restrict__ z, const float* __restrict__ inv_nrm,
                          float* __restrict__ o_attn, float* __restrict__ o_attn_raw,
                          bhalf* __restrict__ attn_b) {
    int row  = blockIdx.x * (blockDim.x / 32) + (threadIdx.x / 32); // b*256 + o2
    int lane = threadIdx.x & 31;
    if (row >= BSZ * OUTC) return;
    int b = row >> 8;
    const float* zr  = z + (size_t)row * LP;
    const float* inv = inv_nrm + b * LP;
    float a[7], mx = -3.0e38f;
#pragma unroll
    for (int t = 0; t < 7; ++t) {
        int l = lane + 32 * t;
        a[t] = (l < LL) ? zr[1 + l] * inv[1 + l] : -3.0e38f;
        mx = fmaxf(mx, a[t]);
    }
    for (int o = 16; o > 0; o >>= 1) mx = fmaxf(mx, __shfl_xor(mx, o, 32));
    float se = 0.0f;
#pragma unroll
    for (int t = 0; t < 7; ++t) {
        int l = lane + 32 * t;
        a[t] = (l < LL) ? __expf(a[t] - mx) : 0.0f;
        se += a[t];
    }
    for (int o = 16; o > 0; o >>= 1) se += __shfl_xor(se, o, 32);
    float is = 1.0f / se;
#pragma unroll
    for (int t = 0; t < 7; ++t) {
        int l = lane + 32 * t;
        float p = a[t] * is;
        if (l < LL) {
            o_attn[(size_t)row * LL + l]     = p;
            o_attn_raw[(size_t)row * LL + l] = p;
            attn_b[(size_t)row * LP + l]     = (bhalf)p;
        } else {
            attn_b[(size_t)row * LP + l] = (bhalf)0.0f;  // zero K-pad for GEMM3
        }
    }
}

// ---------------- GEMM kernels (1 wave / block, 64x32 tile) ----------------

// h = w1_p x xcat  -> stored transposed f32: h_t[b][l][o]
__global__ void k_gemm1(const bhalf* __restrict__ W, const bhalf* __restrict__ Xt,
                        float* __restrict__ Ht) {
    int m0 = blockIdx.x * 64, n0 = blockIdx.y * 32, b = blockIdx.z;
    v8f acc[4][2];
    wmma_mainloop<4, 2>(W, CCH, m0, Xt + (size_t)b * LP * CCH, CCH, n0, CCH, acc);
    store_t_f32<4, 2>(acc, Ht + (size_t)b * LP * HID, HID, m0, n0);
}

// z = w2_p x relu(bn(h)) -> z[b][o2][l] f32
__global__ void k_gemm2(const bhalf* __restrict__ W, const bhalf* __restrict__ Hb,
                        float* __restrict__ Z) {
    int m0 = blockIdx.x * 64, n0 = blockIdx.y * 32, b = blockIdx.z;
    v8f acc[4][2];
    wmma_mainloop<4, 2>(W, HID, m0, Hb + (size_t)b * LP * HID, HID, n0, HID, acc);
    store_n_f32<4, 2>(acc, Z + (size_t)b * OUTC * LP, LP, m0, n0);
}

// obj_val[b][h] = xh(256xL) x attn^T(Lx32) -> stored transposed bf16: ov_t[b][k][h*256+c]
__global__ void k_gemm3(const bhalf* __restrict__ Xh, const bhalf* __restrict__ At,
                        bhalf* __restrict__ OvT) {
    int m0 = blockIdx.x * 64;
    int bh = blockIdx.y, b = bh >> 3, h = bh & 7;
    const bhalf* A  = Xh + ((size_t)b * CCH + h * 256) * LP;
    const bhalf* Bp = At + ((size_t)b * OUTC + h * KD) * LP;
    v8f acc[4][2];
    wmma_mainloop<4, 2>(A, LP, m0, Bp, LP, 0, LP, acc);
    store_t_bf16<4, 2>(acc, OvT + (size_t)b * KD * HID + h * 256, HID, m0, 0);
}

// h2 = w1_o x obj_val -> stored transposed f32: h2_t[b][k][o]
__global__ void k_gemm4(const bhalf* __restrict__ W, const bhalf* __restrict__ OvT,
                        float* __restrict__ H2t) {
    int m0 = blockIdx.x * 64, b = blockIdx.z;
    v8f acc[4][2];
    wmma_mainloop<4, 2>(W, HID, m0, OvT + (size_t)b * KD * HID, HID, 0, HID, acc);
    store_t_f32<4, 2>(acc, H2t + (size_t)b * KD * HID, HID, m0, 0);
}

// out = w2_o x relu(bn(h2)) -> obj_val[b][256][32] f32 (final output section)
__global__ void k_gemm5(const bhalf* __restrict__ W, const bhalf* __restrict__ Hb2,
                        float* __restrict__ Out) {
    int m0 = blockIdx.x * 64, b = blockIdx.z;
    v8f acc[4][2];
    wmma_mainloop<4, 2>(W, HID, m0, Hb2 + (size_t)b * KD * HID, HID, 0, HID, acc);
    store_n_f32<4, 2>(acc, Out + (size_t)b * OUTC * KD, KD, m0, 0);
}

// ---------------- host launcher ----------------

extern "C" void kernel_launch(void* const* d_in, const int* in_sizes, int n_in,
                              void* d_out, int out_size, void* d_ws, size_t ws_size,
                              hipStream_t stream) {
    const float* x    = (const float*)d_in[0];
    const float* w1_p = (const float*)d_in[1];
    const float* g_p  = (const float*)d_in[2];
    const float* b_p  = (const float*)d_in[3];
    const float* w2_p = (const float*)d_in[4];
    const float* w1_o = (const float*)d_in[5];
    const float* g_o  = (const float*)d_in[6];
    const float* b_o  = (const float*)d_in[7];
    const float* w2_o = (const float*)d_in[8];

    float* out        = (float*)d_out;
    float* o_zg       = out;                                     // 64*256
    float* o_val      = out + BSZ * OUTC;                        // 64*256*32
    float* o_attn     = o_val + (size_t)BSZ * OUTC * KD;         // 64*8*32*196
    float* o_attn_raw = o_attn + (size_t)BSZ * OUTC * LL;

    // workspace carve (256B aligned slabs)
    char* w = (char*)d_ws;
    auto alloc = [&](size_t bytes) { void* p = (void*)w; w += (bytes + 255) & ~(size_t)255; return p; };
    bhalf* w1p_b  = (bhalf*)alloc((size_t)HID * CCH * 2);
    bhalf* w2p_b  = (bhalf*)alloc((size_t)OUTC * HID * 2);
    bhalf* w1o_b  = (bhalf*)alloc((size_t)HID * CCH * 2);
    bhalf* w2o_b  = (bhalf*)alloc((size_t)OUTC * HID * 2);
    float* pool   = (float*)alloc((size_t)BSZ * CCH * 4);
    bhalf* xcat_t = (bhalf*)alloc((size_t)BSZ * LP * CCH * 2);
    bhalf* xh_b   = (bhalf*)alloc((size_t)BSZ * CCH * LP * 2);
    float* h_t    = (float*)alloc((size_t)BSZ * LP * HID * 4);
    bhalf* hb     = (bhalf*)alloc((size_t)BSZ * LP * HID * 2);
    float* z      = (float*)alloc((size_t)BSZ * OUTC * LP * 4);
    float* invn   = (float*)alloc((size_t)BSZ * LP * 4);
    bhalf* attn_b = (bhalf*)alloc((size_t)BSZ * OUTC * LP * 2);
    bhalf* ov_t   = (bhalf*)alloc((size_t)BSZ * KD * HID * 2);
    float* h2_t   = (float*)alloc((size_t)BSZ * KD * HID * 4);
    bhalf* hb2    = (bhalf*)alloc((size_t)BSZ * KD * HID * 2);
    float* bn1    = (float*)alloc(4096 * 4);   // sum1 | sq1
    float* bn2    = (float*)alloc(4096 * 4);   // sum2 | sq2

    // 1) weight conversion to bf16
    {
        int n1 = HID * CCH, n2 = OUTC * HID;
        k_f32_to_bf16<<<(n1 + 255) / 256, 256, 0, stream>>>(w1_p, w1p_b, n1);
        k_f32_to_bf16<<<(n2 + 255) / 256, 256, 0, stream>>>(w2_p, w2p_b, n2);
        k_f32_to_bf16<<<(n1 + 255) / 256, 256, 0, stream>>>(w1_o, w1o_b, n1);
        k_f32_to_bf16<<<(n2 + 255) / 256, 256, 0, stream>>>(w2_o, w2o_b, n2);
    }
    // 2) pool + xh_b (bf16, L padded), then K-major xcat_t via LDS transpose
    k_pool_xhb<<<(BSZ * CCH) / 8, 256, 0, stream>>>(x, pool, xh_b);
    k_xcat_t<<<dim3(LP / 32, CCH / 32, BSZ), dim3(32, 8), 0, stream>>>(x, pool, xcat_t);
    // 3) GEMM1 -> h_t[b][l][o]
    k_gemm1<<<dim3(HID / 64, LP / 32, BSZ), 32, 0, stream>>>(w1p_b, xcat_t, h_t);
    // 4) BN stats (over b, l<197) + BN/ReLU -> hb bf16
    k_zero<<<16, 256, 0, stream>>>(bn1, 4096);
    k_bnreduce<<<dim3(8, BSZ), 256, 0, stream>>>(h_t, 197, LP, bn1, bn1 + 2048);
    {
        int total = BSZ * LP * HID;
        k_bnrelu<<<(total + 255) / 256, 256, 0, stream>>>(h_t, bn1, bn1 + 2048, g_p, b_p,
                                                          1.0f / (BSZ * 197), total, hb);
    }
    // 5) GEMM2 -> z[b][o2][l]
    k_gemm2<<<dim3(OUTC / 64, LP / 32, BSZ), 32, 0, stream>>>(w2p_b, hb, z);
    // 6) z_g, channel L2 norm, softmax (-> d_out attn sections + bf16 attn for GEMM3)
    k_zg<<<(BSZ * OUTC) / 256, 256, 0, stream>>>(z, o_zg);
    k_norm<<<dim3(LP / 32, BSZ), dim3(32, 8), 0, stream>>>(z, invn);
    k_softmax<<<(BSZ * OUTC) / 8, 256, 0, stream>>>(z, invn, o_attn, o_attn_raw, attn_b);
    // 7) GEMM3 per (b,head): xh x attn^T -> ov_t[b][k][c] bf16
    k_gemm3<<<dim3(256 / 64, BSZ * NH), 32, 0, stream>>>(xh_b, attn_b, ov_t);
    // 8) second MLP
    k_gemm4<<<dim3(HID / 64, 1, BSZ), 32, 0, stream>>>(w1o_b, ov_t, h2_t);
    k_zero<<<16, 256, 0, stream>>>(bn2, 4096);
    k_bnreduce<<<dim3(8, BSZ), 256, 0, stream>>>(h2_t, KD, KD, bn2, bn2 + 2048);
    {
        int total = BSZ * KD * HID;
        k_bnrelu<<<(total + 255) / 256, 256, 0, stream>>>(h2_t, bn2, bn2 + 2048, g_o, b_o,
                                                          1.0f / (BSZ * KD), total, hb2);
    }
    k_gemm5<<<dim3(OUTC / 64, 1, BSZ), 32, 0, stream>>>(w2o_b, hb2, o_val);

    (void)in_sizes; (void)n_in; (void)out_size; (void)ws_size;
}
